// InteractionNetwork_5763846111244
// MI455X (gfx1250) — matrix operand order repaired and checked by reading, not compiled, not run on hardware
//
#include <hip/hip_runtime.h>
#include <hip/hip_bf16.h>

// ---------------------------------------------------------------------------
// InteractionNetwork on gfx1250 (MI455X): all GEMMs via v_wmma_f32_16x16x32_f16,
// one-time layout transposes so every LDS staging step is b128-vectorized,
// W tiles staged through the Tensor Data Mover (tensor_load_to_lds) when the
// toolchain exposes the builtin.
// B=8, N=1024, R=16384, D_OBJ=32, D_EFF=64, HID=512.
// ---------------------------------------------------------------------------

typedef __attribute__((ext_vector_type(16))) _Float16 v16h;
typedef __attribute__((ext_vector_type(8)))  float    v8f;
typedef __attribute__((ext_vector_type(4)))  unsigned int u32x4;
typedef __attribute__((ext_vector_type(8)))  int      i32x8;
typedef __attribute__((ext_vector_type(4)))  int      i32x4;

static_assert(sizeof(v16h) == 32, "v16h must be 32 bytes");

#define LDA 48  // LDS stride (halves) for manual-staged tiles: 96B, 16B-multiple
#define LDW 32  // LDS stride (halves) for TDM-staged W tiles: packed

#if defined(__has_builtin)
#if __has_builtin(__builtin_amdgcn_tensor_load_to_lds)
#define HAVE_TDM 1
#endif
#if __has_builtin(__builtin_amdgcn_s_wait_tensorcnt)
#define HAVE_TENSORCNT 1
#endif
#endif

__device__ inline v8f wmma_f16(v16h a, v16h b, v8f c) {
    return __builtin_amdgcn_wmma_f32_16x16x32_f16(
        false, a, false, b, (short)0, c, false, false);
}

// A fragment (16x32 MxK) from LDS row-major tile; kbase = lane<16 ? 0 : 8.
__device__ inline v16h load_fragA(const _Float16* rowbase, int kbase) {
    union { v16h v; float4 f[2]; } u;
    u.f[0] = *(const float4*)(rowbase + kbase);
    u.f[1] = *(const float4*)(rowbase + kbase + 16);
    return u.v;
}
// B fragment (32x16 KxN) from LDS K-contiguous-per-column tile; kbase = lane<16?0:8.
__device__ inline v16h load_fragB(const _Float16* colbase, int kbase) {
    union { v16h v; float4 f[2]; } u;
    u.f[0] = *(const float4*)(colbase + kbase);
    u.f[1] = *(const float4*)(colbase + kbase + 16);
    return u.v;
}

#ifdef HAVE_TDM
// Issue a TDM 2D tile load (32 x ncols halves, row stride K halves) into LDS.
__device__ inline void tdm_load_w_tile(const _Float16* gsrc, unsigned lds_addr,
                                       int K, int ncols) {
    unsigned long long ga = (unsigned long long)(size_t)gsrc;
    u32x4 g0;
    g0[0] = 1u;                                    // count=1 (user descriptor)
    g0[1] = lds_addr;                              // LDS byte address
    g0[2] = (unsigned)ga;                          // global_addr[31:0]
    g0[3] = (unsigned)(ga >> 32) | (2u << 30);     // global_addr[56:32] | type=2
    i32x8 g1;
    g1[0] = (1 << 16);                             // workgroup_mask=0, data_size=2B
    g1[1] = (32 << 16);                            // tensor_dim0[15:0]=32 (bits63:48)
    g1[2] = (ncols << 16);                         // dim0 hi=0 | tensor_dim1 lo=ncols
    g1[3] = (32 << 16);                            // dim1 hi=0 | tile_dim0=32
    g1[4] = ncols;                                 // tile_dim1=ncols, tile_dim2=0
    g1[5] = K;                                     // tensor_dim0_stride[31:0]
    g1[6] = 0;
    g1[7] = 0;
    i32x4 z4 = {0, 0, 0, 0};
#if __clang_major__ >= 23
    i32x8 z8 = {0, 0, 0, 0, 0, 0, 0, 0};
    __builtin_amdgcn_tensor_load_to_lds(g0, g1, z4, z4, z8, 0);
#else
    __builtin_amdgcn_tensor_load_to_lds(g0, g1, z4, z4, 0);
#endif
#ifdef HAVE_TENSORCNT
    __builtin_amdgcn_s_wait_tensorcnt(0);
#else
    asm volatile("s_wait_tensorcnt 0x0" ::: "memory");
#endif
}
#endif

// ---------------------------------------------------------------------------
// Prep kernels: converts / transposes (one-time, bandwidth-bound)
// ---------------------------------------------------------------------------
__global__ void cvt_f32_f16(const float* __restrict__ src,
                            _Float16* __restrict__ dst, size_t n) {
    size_t i = (size_t)blockIdx.x * blockDim.x + threadIdx.x;
    size_t stride = (size_t)gridDim.x * blockDim.x;
    for (; i < n; i += stride) dst[i] = (_Float16)src[i];
}

// src f32 [M,C] -> dst f16 [C,M], 64x64 tiles. grid=(C/64, M/64, batch)
__global__ __launch_bounds__(256)
void transpose_f32_f16(const float* __restrict__ src, _Float16* __restrict__ dst,
                       int M, int C) {
    __shared__ _Float16 T[64 * 65];
    size_t nb = (size_t)blockIdx.z * M * C;
    int c0 = blockIdx.x * 64, m0 = blockIdx.y * 64;
    int t = threadIdx.x;
    for (int i = 0; i < 16; ++i) {
        int idx = t + i * 256;
        int mm = idx >> 6, cc = idx & 63;
        T[mm * 65 + cc] = (_Float16)src[nb + (size_t)(m0 + mm) * C + c0 + cc];
    }
    __syncthreads();
    for (int i = 0; i < 16; ++i) {
        int idx = t + i * 256;
        int cc = idx >> 6, mm = idx & 63;
        dst[nb + (size_t)(c0 + cc) * M + m0 + mm] = T[mm * 65 + cc];
    }
}

// src f16 [M,C] -> dst f16 [C,M], 64x64 tiles. grid=(C/64, M/64, batch)
__global__ __launch_bounds__(256)
void transpose_f16(const _Float16* __restrict__ src, _Float16* __restrict__ dst,
                   int M, int C) {
    __shared__ _Float16 T[64 * 65];
    size_t nb = (size_t)blockIdx.z * M * C;
    int c0 = blockIdx.x * 64, m0 = blockIdx.y * 64;
    int t = threadIdx.x;
    for (int i = 0; i < 16; ++i) {
        int idx = t + i * 256;
        int mm = idx >> 6, cc = idx & 63;
        T[mm * 65 + cc] = src[nb + (size_t)(m0 + mm) * C + c0 + cc];
    }
    __syncthreads();
    for (int i = 0; i < 16; ++i) {
        int idx = t + i * 256;
        int cc = idx >> 6, mm = idx & 63;
        dst[nb + (size_t)(c0 + cc) * M + m0 + mm] = T[mm * 65 + cc];
    }
}

// WT[c*K + k] = (f16) W[k*N + c]   (weights are tiny; scalar is fine)
__global__ void cvt_transpose_w(const float* __restrict__ W,
                                _Float16* __restrict__ WT, int K, int N) {
    int i = blockIdx.x * blockDim.x + threadIdx.x;
    if (i >= K * N) return;
    int c = i / K, k = i % K;
    WT[i] = (_Float16)W[(size_t)k * N + c];
}

// objT[(b*32+c)*N + n] = (f16) obj[(b*N+n)*32 + c]
__global__ void build_objT(const float* __restrict__ obj,
                           _Float16* __restrict__ objT) {
    const int N = 1024, D = 32;
    int i = blockIdx.x * blockDim.x + threadIdx.x;   // B*D*N total
    int b = i / (D * N), rem = i % (D * N);
    int c = rem / N, n = rem % N;
    objT[i] = (_Float16)obj[((size_t)b * N + n) * D + c];
}

// update[:, 0:32] = obj (f16)
__global__ void concat_obj(const float* __restrict__ obj,
                           _Float16* __restrict__ update) {
    int i = blockIdx.x * blockDim.x + threadIdx.x;   // 262144
    int row = i >> 5, c = i & 31;
    update[(size_t)row * 96 + c] = (_Float16)obj[i];
}

// ---------------------------------------------------------------------------
// Gather: node[b,r,0:32] = Rs^T obj ; node[b,r,32:64] = Rr^T obj
// A from RsT/RrT [R,N] (K=n contiguous). grid=(R/128, B), 256 threads.
// ---------------------------------------------------------------------------
__global__ __launch_bounds__(256)
void gather_gemm(const _Float16* __restrict__ RsT, const _Float16* __restrict__ RrT,
                 const _Float16* __restrict__ objT, _Float16* __restrict__ node) {
    __shared__ _Float16 Ss[128 * LDA];
    __shared__ _Float16 Ds[128 * LDA];
    __shared__ _Float16 Bs[32 * LDA];
    const int Rtot = 16384, Ntot = 1024;
    int t = threadIdx.x, lane = t & 31, wave = t >> 5;
    int r0 = blockIdx.x * 128;
    int b  = blockIdx.y;

    v8f acc[4] = {v8f{}, v8f{}, v8f{}, v8f{}};
    for (int k0 = 0; k0 < Ntot; k0 += 32) {
        __syncthreads();
        for (int i = 0; i < 2; ++i) {               // 512 chunks of 8 halves
            int idx = t + i * 256;
            int r = idx >> 2, kseg = (idx & 3) * 8;
            *(float4*)&Ss[r * LDA + kseg] =
                *(const float4*)&RsT[(size_t)(r0 + r) * Ntot + k0 + kseg];
            *(float4*)&Ds[r * LDA + kseg] =
                *(const float4*)&RrT[(size_t)(r0 + r) * Ntot + k0 + kseg];
        }
        if (t < 128) {                              // 32x32 obj tile
            int c = t >> 2, kseg = (t & 3) * 8;
            *(float4*)&Bs[c * LDA + kseg] =
                *(const float4*)&objT[((size_t)b * 32 + c) * Ntot + k0 + kseg];
        }
        __syncthreads();
        int arow = wave * 16 + (lane & 15);
        int kbA = (lane < 16) ? 0 : 8;
        v16h as = load_fragA(&Ss[arow * LDA], kbA);
        v16h ad = load_fragA(&Ds[arow * LDA], kbA);
        int kbB = (lane < 16) ? 0 : 8;
        v16h b0 = load_fragB(&Bs[(lane & 15) * LDA], kbB);
        v16h b1 = load_fragB(&Bs[(16 + (lane & 15)) * LDA], kbB);
        acc[0] = wmma_f16(as, b0, acc[0]);
        acc[1] = wmma_f16(as, b1, acc[1]);
        acc[2] = wmma_f16(ad, b0, acc[2]);
        acc[3] = wmma_f16(ad, b1, acc[3]);
    }
    int rbase = r0 + wave * 16 + ((lane >> 4) << 3);
    for (int ct = 0; ct < 4; ++ct) {
        int col = (ct & 1) * 16 + (lane & 15) + ((ct >= 2) ? 32 : 0);
        for (int j = 0; j < 8; ++j)
            node[((size_t)b * Rtot + rbase + j) * 64 + col] = (_Float16)acc[ct][j];
    }
}

// ---------------------------------------------------------------------------
// Generic GEMM: out = act(A[M,K] @ W + bias), W given pre-transposed WT[N,K].
// Tile 256x64, 8 waves x 32 rows. grid=(M/256, ceil(N/64)).
// W tile staged by TDM (tensor_load_to_lds) when available.
// ---------------------------------------------------------------------------
template <bool RELU, bool OUT_F32>
__global__ __launch_bounds__(256)
void gemm_f16(const _Float16* __restrict__ A, const _Float16* __restrict__ WT,
              const float* __restrict__ bias, void* __restrict__ outv,
              int M, int K, int N) {
    __shared__ _Float16 As[256 * LDA];
    __shared__ _Float16 Ws[64 * LDW];
    int t = threadIdx.x, lane = t & 31, wave = t >> 5;
    int row0 = blockIdx.x * 256;
    int col0 = blockIdx.y * 64;
    int ncols = N - col0; if (ncols > 64) ncols = 64;
    int nct = ncols >> 4;

    v8f acc[2][4] = {{v8f{}, v8f{}, v8f{}, v8f{}}, {v8f{}, v8f{}, v8f{}, v8f{}}};
    for (int k0 = 0; k0 < K; k0 += 32) {
        __syncthreads();
        // A tile 256x32, b128 chunks
        for (int i = 0; i < 4; ++i) {
            int idx = t + i * 256;                  // 1024 chunks
            int r = idx >> 2, kseg = (idx & 3) * 8;
            *(float4*)&As[r * LDA + kseg] =
                *(const float4*)&A[(size_t)(row0 + r) * K + k0 + kseg];
        }
#ifdef HAVE_TDM
        if (wave == 0) {
            tdm_load_w_tile(&WT[(size_t)col0 * K + k0],
                            (unsigned)(size_t)(void*)Ws, K, ncols);
        }
#else
        if (t < 256) {
            int idx = t;                            // 256 chunks (64c x 32k /8)
            int c = idx >> 2, kseg = (idx & 3) * 8;
            float4 v = {0.f, 0.f, 0.f, 0.f};
            if (c < ncols)
                v = *(const float4*)&WT[(size_t)(col0 + c) * K + k0 + kseg];
            *(float4*)&Ws[c * LDW + kseg] = v;
        }
#endif
        __syncthreads();
        int arow = wave * 32 + (lane & 15);
        int kbA = (lane < 16) ? 0 : 8;
        v16h a0 = load_fragA(&As[arow * LDA], kbA);
        v16h a1 = load_fragA(&As[(arow + 16) * LDA], kbA);
        int kbB = (lane < 16) ? 0 : 8;
        for (int ct = 0; ct < nct; ++ct) {
            v16h bb = load_fragB(&Ws[(ct * 16 + (lane & 15)) * LDW], kbB);
            acc[0][ct] = wmma_f16(a0, bb, acc[0][ct]);
            acc[1][ct] = wmma_f16(a1, bb, acc[1][ct]);
        }
    }
    for (int p = 0; p < 2; ++p) {
        int rbase = row0 + wave * 32 + p * 16 + ((lane >> 4) << 3);
        for (int ct = 0; ct < nct; ++ct) {
            int col = col0 + ct * 16 + (lane & 15);
            float bv = bias ? bias[col] : 0.f;
            for (int j = 0; j < 8; ++j) {
                float v = acc[p][ct][j] + bv;
                if (RELU) v = fmaxf(v, 0.f);
                size_t off = (size_t)(rbase + j) * N + col;
                if (OUT_F32) ((float*)outv)[off] = v;
                else         ((_Float16*)outv)[off] = (_Float16)v;
            }
        }
    }
}

// ---------------------------------------------------------------------------
// Scatter: update[b,n,32+e] = sum_r Rr[n,r] * edge[b,r,e]
// A = Rr f16 [N,R] (K contiguous), B from edgeT [B,64,R]. grid=(N/128, B).
// ---------------------------------------------------------------------------
__global__ __launch_bounds__(256)
void scatter_gemm(const _Float16* __restrict__ Rrh, const _Float16* __restrict__ edgeT,
                  _Float16* __restrict__ update) {
    __shared__ _Float16 As[128 * LDA];
    __shared__ _Float16 Bs[64 * LDA];
    const int Rtot = 16384, Ntot = 1024;
    int t = threadIdx.x, lane = t & 31, wave = t >> 5;
    int n0 = blockIdx.x * 128;
    int b  = blockIdx.y;

    v8f acc[4] = {v8f{}, v8f{}, v8f{}, v8f{}};
    for (int k0 = 0; k0 < Rtot; k0 += 32) {
        __syncthreads();
        for (int i = 0; i < 2; ++i) {               // 512 chunks
            int idx = t + i * 256;
            int r = idx >> 2, kseg = (idx & 3) * 8;
            *(float4*)&As[r * LDA + kseg] =
                *(const float4*)&Rrh[(size_t)(n0 + r) * Rtot + k0 + kseg];
        }
        {
            int c = t >> 2, kseg = (t & 3) * 8;     // 256 chunks (64c x 32k /8)
            *(float4*)&Bs[c * LDA + kseg] =
                *(const float4*)&edgeT[((size_t)b * 64 + c) * Rtot + k0 + kseg];
        }
        __syncthreads();
        int arow = wave * 16 + (lane & 15);
        v16h a = load_fragA(&As[arow * LDA], (lane < 16) ? 0 : 8);
        int kbB = (lane < 16) ? 0 : 8;
        for (int ct = 0; ct < 4; ++ct) {
            v16h bb = load_fragB(&Bs[(ct * 16 + (lane & 15)) * LDA], kbB);
            acc[ct] = wmma_f16(a, bb, acc[ct]);
        }
    }
    int rbase = n0 + wave * 16 + ((lane >> 4) << 3);
    for (int ct = 0; ct < 4; ++ct) {
        int col = ct * 16 + (lane & 15);
        for (int j = 0; j < 8; ++j)
            update[((size_t)b * Ntot + rbase + j) * 96 + 32 + col] = (_Float16)acc[ct][j];
    }
}

// ---------------------------------------------------------------------------
// Host launcher
// ---------------------------------------------------------------------------
extern "C" void kernel_launch(void* const* d_in, const int* in_sizes, int n_in,
                              void* d_out, int out_size, void* d_ws, size_t ws_size,
                              hipStream_t stream) {
    (void)in_sizes; (void)n_in; (void)out_size; (void)ws_size;
    const int B = 8, N = 1024, R = 16384, D = 32, E = 64, H = 512;

    const float* obj = (const float*)d_in[0];
    const float* Rs  = (const float*)d_in[1];
    const float* Rr  = (const float*)d_in[2];
    const float* rw1 = (const float*)d_in[3];  const float* rb1 = (const float*)d_in[4];
    const float* rw2 = (const float*)d_in[5];  const float* rb2 = (const float*)d_in[6];
    const float* rw3 = (const float*)d_in[7];  const float* rb3 = (const float*)d_in[8];
    const float* rw4 = (const float*)d_in[9];  const float* rb4 = (const float*)d_in[10];
    const float* ow1 = (const float*)d_in[11]; const float* ob1 = (const float*)d_in[12];
    const float* ow2 = (const float*)d_in[13]; const float* ob2 = (const float*)d_in[14];

    char* base = (char*)d_ws;
    size_t off = 0;
    auto alloc = [&](size_t nhalf) {
        char* p = base + off;
        off += (nhalf * sizeof(_Float16) + 255) & ~(size_t)255;
        return (_Float16*)p;
    };
    _Float16* RsT   = alloc((size_t)R * N);           // [R,N]
    _Float16* RrT   = alloc((size_t)R * N);           // [R,N]
    _Float16* Rrh   = alloc((size_t)N * R);           // [N,R]
    _Float16* objT  = alloc((size_t)B * D * N);       // [B,32,N]
    _Float16* rw1t  = alloc((size_t)H * 2 * D);       // [512,64]
    _Float16* rw2t  = alloc((size_t)H * H);           // [512,512]
    _Float16* rw3t  = alloc((size_t)E * H);           // [64,512]
    _Float16* rw4t  = alloc((size_t)E * E);           // [64,64]
    _Float16* ow1t  = alloc((size_t)H * (D + E));     // [512,96]
    _Float16* ow2t  = alloc((size_t)D * H);           // [32,512]
    _Float16* node  = alloc((size_t)B * R * 64);      // [BR,64]
    _Float16* h1    = alloc((size_t)B * R * H);       // [BR,512]
    _Float16* h2    = alloc((size_t)B * R * H);       // [BR,512]
    _Float16* h3    = alloc((size_t)B * R * E);       // [BR,64]
    _Float16* edge  = alloc((size_t)B * R * E);       // [BR,64]
    _Float16* edgeT = alloc((size_t)B * E * R);       // [B,64,R]
    _Float16* upd   = alloc((size_t)B * N * (D + E)); // [BN,96]
    _Float16* h4    = alloc((size_t)B * N * H);       // [BN,512]

    // ---- one-time layout prep ----
    transpose_f32_f16<<<dim3(R / 64, N / 64, 1), 256, 0, stream>>>(Rs, RsT, N, R);
    transpose_f32_f16<<<dim3(R / 64, N / 64, 1), 256, 0, stream>>>(Rr, RrT, N, R);
    {
        size_t n = (size_t)N * R;
        cvt_f32_f16<<<8192, 256, 0, stream>>>(Rr, Rrh, n);
    }
    auto cvtw = [&](const float* w, _Float16* wt, int K, int Nn) {
        int tot = K * Nn;
        cvt_transpose_w<<<(tot + 255) / 256, 256, 0, stream>>>(w, wt, K, Nn);
    };
    cvtw(rw1, rw1t, 2 * D, H);
    cvtw(rw2, rw2t, H, H);
    cvtw(rw3, rw3t, H, E);
    cvtw(rw4, rw4t, E, E);
    cvtw(ow1, ow1t, D + E, H);
    cvtw(ow2, ow2t, H, D);
    build_objT<<<(B * D * N) / 256, 256, 0, stream>>>(obj, objT);
    concat_obj<<<(B * N * D) / 256, 256, 0, stream>>>(obj, upd);

    // ---- gather ----
    gather_gemm<<<dim3(R / 128, B), 256, 0, stream>>>(RsT, RrT, objT, node);

    const int BR = B * R, BN = B * N;
    // ---- relational MLP ----
    gemm_f16<true, false><<<dim3(BR / 256, H / 64), 256, 0, stream>>>(
        node, rw1t, rb1, h1, BR, 2 * D, H);
    gemm_f16<true, false><<<dim3(BR / 256, H / 64), 256, 0, stream>>>(
        h1, rw2t, rb2, h2, BR, H, H);
    gemm_f16<true, false><<<dim3(BR / 256, E / 64), 256, 0, stream>>>(
        h2, rw3t, rb3, h3, BR, H, E);
    gemm_f16<true, false><<<dim3(BR / 256, E / 64), 256, 0, stream>>>(
        h3, rw4t, rb4, edge, BR, E, E);

    // ---- scatter-aggregate ----
    transpose_f16<<<dim3(1, R / 64, B), 256, 0, stream>>>(edge, edgeT, R, E);
    scatter_gemm<<<dim3(N / 128, B), 256, 0, stream>>>(Rrh, edgeT, upd);

    // ---- object MLP ----
    gemm_f16<true, false><<<dim3(BN / 256, H / 64), 256, 0, stream>>>(
        upd, ow1t, ob1, h4, BN, D + E, H);
    gemm_f16<false, true><<<dim3(BN / 256, 1), 256, 0, stream>>>(
        h4, ow2t, ob2, d_out, BN, H, D);
}
